// EncoderLayer_59442347377602
// MI455X (gfx1250) — matrix operand order repaired and checked by reading, not compiled, not run on hardware
//
#include <hip/hip_runtime.h>
#include <math.h>

#define DIMV   1024
#define NHEAD  16
#define HD     64
#define FFD    4096
#define NB     2
#define SEQ    2048
#define RTOT   (NB*SEQ)          // 4096 rows total
#define NEGF   (-3.4028234663852886e38f)

typedef _Float16 v16h __attribute__((ext_vector_type(16)));
typedef float    v8f  __attribute__((ext_vector_type(8)));
typedef float    f4   __attribute__((ext_vector_type(4)));

// ---------------------------------------------------------------- WMMA helpers

__device__ __forceinline__ v8f wmma_f16(v16h a, v16h b, v8f c) {
  // D = A(16x32 f16) * B(32x16 f16) + C(16x16 f32)
  return __builtin_amdgcn_wmma_f32_16x16x32_f16(false, a, false, b, (short)0, c, false, false);
}

// A fragment (16x32): lane lo = M row; element e -> K = kbase + 8*hi + e + (e>=8 ? 8 : 0)
__device__ __forceinline__ v16h load_a(const _Float16* p, int stride, int row, int kbase, int hi) {
  const _Float16* base = p + (size_t)row * stride + kbase + hi * 8;
  union { v16h v; f4 f[2]; } u;
  u.f[0] = *reinterpret_cast<const f4*>(base);
  u.f[1] = *reinterpret_cast<const f4*>(base + 16);
  return u.v;
}

// B fragment (32x16) read via Bt[N][K]: lane lo = N col; element e -> K = kbase + 8*hi + e + (e>=8 ? 8 : 0)
__device__ __forceinline__ v16h load_bt(const _Float16* p, int stride, int n, int kbase, int hi) {
  const _Float16* base = p + (size_t)n * stride + kbase + hi * 8;
  union { v16h v; f4 f[2]; } u;
  u.f[0] = *reinterpret_cast<const f4*>(base);
  u.f[1] = *reinterpret_cast<const f4*>(base + 16);
  return u.v;
}

// ------------------------------------------------------- DPP 16-lane reductions
// Reductions live entirely inside each 16-lane half-wave (== one DPP row),
// so use pure-VALU DPP butterflies instead of ds_bpermute round-trips.

template <int CTRL>
__device__ __forceinline__ float dpp_mv(float v) {
  int y = __builtin_amdgcn_update_dpp(__float_as_int(v), __float_as_int(v),
                                      CTRL, 0xF, 0xF, true);
  return __int_as_float(y);
}
__device__ __forceinline__ float red_max16(float v) {
  v = fmaxf(v, dpp_mv<0xB1>(v));    // quad_perm [1,0,3,2]  (xor 1)
  v = fmaxf(v, dpp_mv<0x4E>(v));    // quad_perm [2,3,0,1]  (xor 2)
  v = fmaxf(v, dpp_mv<0x141>(v));   // row_half_mirror      (covers 8)
  v = fmaxf(v, dpp_mv<0x140>(v));   // row_mirror           (covers 16)
  return v;
}
__device__ __forceinline__ float red_sum16(float v) {
  v += dpp_mv<0xB1>(v);
  v += dpp_mv<0x4E>(v);
  v += dpp_mv<0x141>(v);
  v += dpp_mv<0x140>(v);
  return v;
}

// ---------------------------------------------------------------- small kernels

// f32 [K x N] row-major  ->  f16 [N x K] row-major (transposed, for WMMA B operand)
__global__ void transpose_to_f16(const float* __restrict__ in, _Float16* __restrict__ out,
                                 int Kd, int Nd) {
  __shared__ float tile[32][33];
  int n0 = blockIdx.x * 32, k0 = blockIdx.y * 32;
  int tx = threadIdx.x, ty = threadIdx.y;
  for (int i = ty; i < 32; i += 8)
    tile[i][tx] = in[(size_t)(k0 + i) * Nd + n0 + tx];
  __syncthreads();
  for (int i = ty; i < 32; i += 8)
    out[(size_t)(n0 + i) * Kd + k0 + tx] = (_Float16)tile[tx][i];
}

// rmsnorm over last dim (DIMV), output f16; one block (256 thr) per row
__global__ void rmsnorm_f16(const float* __restrict__ x, const float* __restrict__ g,
                            _Float16* __restrict__ out) {
  __shared__ float red[256];
  int row = blockIdx.x;
  const float* xr = x + (size_t)row * DIMV;
  float v[4], s = 0.f;
#pragma unroll
  for (int i = 0; i < 4; ++i) { v[i] = xr[threadIdx.x + i * 256]; s += v[i] * v[i]; }
  red[threadIdx.x] = s;
  __syncthreads();
  for (int off = 128; off > 0; off >>= 1) {
    if (threadIdx.x < off) red[threadIdx.x] += red[threadIdx.x + off];
    __syncthreads();
  }
  float inv = rsqrtf(red[0] * (1.0f / DIMV) + 1e-6f);
#pragma unroll
  for (int i = 0; i < 4; ++i) {
    int c = threadIdx.x + i * 256;
    out[(size_t)row * DIMV + c] = (_Float16)(v[i] * inv * g[c]);
  }
}

// ---------------------------------------------------------------- generic WMMA GEMM
// C[row, col] = sum_k A[row,k] * Bt[col,k];  1 wave = 32(M) x 64(N), 4 waves/block.
// B fragments are reused across two A fragments (8 WMMAs per 6KB fetched).

constexpr int M_QHEAD   = 0;  // f16 out scattered to [B,H,S,HD]
constexpr int M_VT      = 1;  // f16 out scattered to [B,H,HD,S]
constexpr int M_F32RES  = 2;  // f32 out = acc + resid
constexpr int M_GELU    = 3;  // f16 out = gelu(acc + bias)
constexpr int M_BIASRES = 4;  // f32 out = acc + bias + resid

template <int MODE, int KD, int ND>
__global__ __launch_bounds__(128)
void gemm_wmma(const _Float16* __restrict__ A, const _Float16* __restrict__ Bt,
               const float* __restrict__ bias, const float* __restrict__ resid,
               void* __restrict__ outp) {
  int wave = threadIdx.x >> 5, lane = threadIdx.x & 31;
  int lo = lane & 15, hi = lane >> 4;
  int rowBase = blockIdx.y * 128 + wave * 32;
  int colBase = blockIdx.x * 64;

  v8f c[2][4] = {};
  for (int k0 = 0; k0 < KD; k0 += 32) {
    v16h a0 = load_a(A, KD, rowBase + lo,      k0, hi);
    v16h a1 = load_a(A, KD, rowBase + 16 + lo, k0, hi);
#pragma unroll
    for (int t = 0; t < 4; ++t) {
      v16h b = load_bt(Bt, KD, colBase + t * 16 + lo, k0, hi);
      c[0][t] = wmma_f16(a0, b, c[0][t]);
      c[1][t] = wmma_f16(a1, b, c[1][t]);
    }
  }

#pragma unroll
  for (int mh = 0; mh < 2; ++mh) {
#pragma unroll
    for (int t = 0; t < 4; ++t) {
#pragma unroll
      for (int r = 0; r < 8; ++r) {
        int row = rowBase + mh * 16 + r + 8 * hi;  // global row in [0, RTOT)
        int col = colBase + t * 16 + lo;           // global col in [0, ND)
        float v = c[mh][t][r];
        if constexpr (MODE == M_QHEAD) {
          int bb = row >> 11, ss = row & (SEQ - 1);
          int hh = col >> 6,  dd = col & (HD - 1);
          ((_Float16*)outp)[((((size_t)bb * NHEAD + hh) * SEQ + ss) << 6) + dd] = (_Float16)v;
        } else if constexpr (MODE == M_VT) {
          int bb = row >> 11, ss = row & (SEQ - 1);
          int hh = col >> 6,  dd = col & (HD - 1);
          ((_Float16*)outp)[(((size_t)bb * NHEAD + hh) * HD + dd) * SEQ + ss] = (_Float16)v;
        } else if constexpr (MODE == M_F32RES) {
          size_t idx = (size_t)row * ND + col;
          ((float*)outp)[idx] = v + resid[idx];
        } else if constexpr (MODE == M_GELU) {
          v += bias[col];
          v = 0.5f * v * (1.0f + erff(v * 0.70710678118654752f));   // exact erf GELU
          ((_Float16*)outp)[(size_t)row * ND + col] = (_Float16)v;
        } else {  // M_BIASRES
          size_t idx = (size_t)row * ND + col;
          ((float*)outp)[idx] = v + bias[col] + resid[idx];
        }
      }
    }
  }
}

// ---------------------------------------------------------------- flash attention
// 1 wave owns a 16-row Q tile; iterates keys 32 at a time with online softmax.
// q,k: [B,H,S,HD] f16.  vt: [B,H,HD,S] f16.  ctx out: [B,S,DIM] f16.

__global__ __launch_bounds__(128)
void attn_wmma(const _Float16* __restrict__ q, const _Float16* __restrict__ k,
               const _Float16* __restrict__ vt, const float* __restrict__ rbias,
               const unsigned char* __restrict__ mask, _Float16* __restrict__ ctx) {
  __shared__ __align__(16) _Float16 plds[4][16][40];   // per-wave P tile (16x32, padded)

  int wave = threadIdx.x >> 5, lane = threadIdx.x & 31;
  int lo = lane & 15, hi = lane >> 4;
  int qt = blockIdx.x * 4 + wave;          // 16-row q tile index, [0,128)
  int h  = blockIdx.y, b = blockIdx.z;
  int qrow = qt * 16;

  const _Float16* qp = q  + ((size_t)b * NHEAD + h) * SEQ * HD;
  const _Float16* kp = k  + ((size_t)b * NHEAD + h) * SEQ * HD;
  const _Float16* vp = vt + ((size_t)b * NHEAD + h) * HD * SEQ;
  const float* bp = rbias + ((size_t)h * SEQ + qrow) * SEQ;
  const unsigned char* mp = mask + (size_t)b * SEQ;

  // Q fragments for head-dim chunks [0,32) and [32,64)
  v16h qa0 = load_a(qp, HD, qrow + lo, 0,  hi);
  v16h qa1 = load_a(qp, HD, qrow + lo, 32, hi);

  v8f acc[4] = {};
  float m[8], l[8];
#pragma unroll
  for (int r = 0; r < 8; ++r) { m[r] = -INFINITY; l[r] = 0.f; }
  const float scale = 0.125f;  // HD^-0.5

  for (int k0 = 0; k0 < SEQ; k0 += 32) {
    v8f s[2];
#pragma unroll
    for (int j = 0; j < 2; ++j) {
      v16h kb0 = load_bt(kp, HD, k0 + j * 16 + lo, 0,  hi);
      v16h kb1 = load_bt(kp, HD, k0 + j * 16 + lo, 32, hi);
      v8f z = {};
      z = wmma_f16(qa0, kb0, z);
      z = wmma_f16(qa1, kb1, z);
      s[j] = z;
    }
    // scale + rel-pos bias + key padding mask, then online softmax update
#pragma unroll
    for (int r = 0; r < 8; ++r) {
      int row = r + 8 * hi;
      float rv = -INFINITY;
#pragma unroll
      for (int j = 0; j < 2; ++j) {
        int key = k0 + j * 16 + lo;
        float v = s[j][r] * scale + bp[(size_t)row * SEQ + key];
        if (mp[key]) v = NEGF;
        s[j][r] = v;
        rv = fmaxf(rv, v);
      }
      rv = red_max16(rv);                       // DPP butterfly, no LDS
      float mnew = fmaxf(m[r], rv);
      float alpha = __expf(m[r] - mnew);
      m[r] = mnew;
      float rs = 0.f;
#pragma unroll
      for (int j = 0; j < 2; ++j) {
        float p = __expf(s[j][r] - mnew);
        s[j][r] = p;
        rs += p;
      }
      rs = red_sum16(rs);                       // DPP butterfly, no LDS
      l[r] = l[r] * alpha + rs;
#pragma unroll
      for (int t = 0; t < 4; ++t) acc[t][r] *= alpha;
    }
    // C-layout -> A-layout transpose of P through LDS (same-wave slice)
#pragma unroll
    for (int r = 0; r < 8; ++r) {
#pragma unroll
      for (int j = 0; j < 2; ++j)
        plds[wave][r + 8 * hi][j * 16 + lo] = (_Float16)s[j][r];
    }
    v16h pa = load_a(&plds[wave][0][0], 40, lo, 0, hi);
#pragma unroll
    for (int t = 0; t < 4; ++t) {
      v16h vb = load_bt(vp, SEQ, t * 16 + lo, k0, hi);
      acc[t] = wmma_f16(pa, vb, acc[t]);
    }
  }

  // normalize and scatter to [B,S,DIM] with head concat
#pragma unroll
  for (int r = 0; r < 8; ++r) {
    float inv = 1.0f / l[r];
    int row = qrow + r + 8 * hi;
#pragma unroll
    for (int t = 0; t < 4; ++t) {
      float o = acc[t][r] * inv;
      ctx[((size_t)b * SEQ + row) * DIMV + h * HD + t * 16 + lo] = (_Float16)o;
    }
  }
}

// ---------------------------------------------------------------- launch

extern "C" void kernel_launch(void* const* d_in, const int* in_sizes, int n_in,
                              void* d_out, int out_size, void* d_ws, size_t ws_size,
                              hipStream_t stream) {
  const float*         x     = (const float*)d_in[0];
  const unsigned char* mask  = (const unsigned char*)d_in[1];   // jax bool -> 1 byte
  const float*         rbias = (const float*)d_in[2];
  const float* Wq = (const float*)d_in[3];
  const float* Wk = (const float*)d_in[4];
  const float* Wv = (const float*)d_in[5];
  const float* Wo = (const float*)d_in[6];
  const float* W1 = (const float*)d_in[7];
  const float* b1 = (const float*)d_in[8];
  const float* W2 = (const float*)d_in[9];
  const float* b2 = (const float*)d_in[10];
  const float* g1 = (const float*)d_in[11];
  const float* g2 = (const float*)d_in[12];
  float* out = (float*)d_out;

  size_t off = 0;
  auto alloc = [&](size_t bytes) -> char* {
    char* p = (char*)d_ws + off;
    off += (bytes + 255) & ~(size_t)255;
    return p;
  };
  _Float16* hA   = (_Float16*)alloc((size_t)RTOT * DIMV * 2);
  _Float16* Wqt  = (_Float16*)alloc((size_t)DIMV * DIMV * 2);
  _Float16* Wkt  = (_Float16*)alloc((size_t)DIMV * DIMV * 2);
  _Float16* Wvt  = (_Float16*)alloc((size_t)DIMV * DIMV * 2);
  _Float16* Wot  = (_Float16*)alloc((size_t)DIMV * DIMV * 2);
  _Float16* W1t  = (_Float16*)alloc((size_t)DIMV * FFD * 2);
  _Float16* W2t  = (_Float16*)alloc((size_t)FFD * DIMV * 2);
  _Float16* qb   = (_Float16*)alloc((size_t)RTOT * DIMV * 2);
  _Float16* kb   = (_Float16*)alloc((size_t)RTOT * DIMV * 2);
  _Float16* vtb  = (_Float16*)alloc((size_t)RTOT * DIMV * 2);
  _Float16* ctx  = (_Float16*)alloc((size_t)RTOT * DIMV * 2);
  float*    x1   = (float*)   alloc((size_t)RTOT * DIMV * 4);
  _Float16* h2   = (_Float16*)alloc((size_t)RTOT * DIMV * 2);
  _Float16* f1   = (_Float16*)alloc((size_t)RTOT * FFD * 2);
  (void)ws_size; (void)n_in; (void)in_sizes; (void)out_size;

  dim3 tb(32, 8);
  transpose_to_f16<<<dim3(DIMV/32, DIMV/32), tb, 0, stream>>>(Wq, Wqt, DIMV, DIMV);
  transpose_to_f16<<<dim3(DIMV/32, DIMV/32), tb, 0, stream>>>(Wk, Wkt, DIMV, DIMV);
  transpose_to_f16<<<dim3(DIMV/32, DIMV/32), tb, 0, stream>>>(Wv, Wvt, DIMV, DIMV);
  transpose_to_f16<<<dim3(DIMV/32, DIMV/32), tb, 0, stream>>>(Wo, Wot, DIMV, DIMV);
  transpose_to_f16<<<dim3(FFD/32,  DIMV/32), tb, 0, stream>>>(W1, W1t, DIMV, FFD);
  transpose_to_f16<<<dim3(DIMV/32, FFD/32),  tb, 0, stream>>>(W2, W2t, FFD, DIMV);

  rmsnorm_f16<<<RTOT, 256, 0, stream>>>(x, g1, hA);

  gemm_wmma<M_QHEAD, DIMV, DIMV><<<dim3(DIMV/64, RTOT/128), 128, 0, stream>>>(hA, Wqt, nullptr, nullptr, qb);
  gemm_wmma<M_QHEAD, DIMV, DIMV><<<dim3(DIMV/64, RTOT/128), 128, 0, stream>>>(hA, Wkt, nullptr, nullptr, kb);
  gemm_wmma<M_VT,    DIMV, DIMV><<<dim3(DIMV/64, RTOT/128), 128, 0, stream>>>(hA, Wvt, nullptr, nullptr, vtb);

  attn_wmma<<<dim3(SEQ/64, NHEAD, NB), 128, 0, stream>>>(qb, kb, vtb, rbias, mask, ctx);

  gemm_wmma<M_F32RES, DIMV, DIMV><<<dim3(DIMV/64, RTOT/128), 128, 0, stream>>>(ctx, Wot, nullptr, x, x1);

  rmsnorm_f16<<<RTOT, 256, 0, stream>>>(x1, g2, h2);

  gemm_wmma<M_GELU,    DIMV, FFD><<<dim3(FFD/64,  RTOT/128), 128, 0, stream>>>(h2, W1t, b1, nullptr, f1);
  gemm_wmma<M_BIASRES, FFD, DIMV><<<dim3(DIMV/64, RTOT/128), 128, 0, stream>>>(f1, W2t, b2, x1, out);
}
